// Block_13288628814487
// MI455X (gfx1250) — matrix operand order repaired and checked by reading, not compiled, not run on hardware
//
#include <hip/hip_runtime.h>
#include <hip/hip_bf16.h>

typedef __attribute__((ext_vector_type(8)))  float   v8f;
typedef __attribute__((ext_vector_type(8)))  __bf16  v8bf;
typedef __attribute__((ext_vector_type(16))) __bf16  v16bf;
typedef __attribute__((__vector_size__(4 * sizeof(int)))) int v4i;

#define N_EMBD 1024
#define N_HEAD 16
#define HEAD_SZ 64
#define BB 2
#define TT 2048
#define BT (BB*TT)
#define FF (4*N_EMBD)

// ---------------- helpers ----------------

__device__ inline v8f wmma_bf16(v16bf a, v16bf b, v8f c) {
    // D = A(16x32 bf16) * B(32x16 bf16) + C(16x16 f32)
    return __builtin_amdgcn_wmma_f32_16x16x32_bf16(
        /*neg_a=*/false, a, /*neg_b=*/false, b,
        /*c_mod=*/(short)0, c, /*reuse_a=*/false, /*reuse_b=*/false);
}

// A-fragment load: elements [0..15]=p[0..15]
__device__ inline v16bf ldA(const __bf16* p) {
    return *(const v16bf*)(p);
}

// B-fragment load: 16 contiguous bf16
__device__ inline v16bf ldB(const __bf16* p) {
    return *(const v16bf*)(p);
}

// CDNA5 async global->LDS copy of 16 bytes per lane (ASYNCcnt-tracked).
__device__ inline void async_b128(const __bf16* gsrc, __bf16* ldst) {
#if __has_builtin(__builtin_amdgcn_global_load_async_to_lds_b128)
    __builtin_amdgcn_global_load_async_to_lds_b128(
        (__attribute__((address_space(1))) v4i*)gsrc,
        (__attribute__((address_space(3))) v4i*)ldst, 0, 0);
#else
    unsigned loff =
        (unsigned)(unsigned long long)(__attribute__((address_space(3))) __bf16*)ldst;
    unsigned long long ga = (unsigned long long)gsrc;
    asm volatile("global_load_async_to_lds_b128 %0, %1, off"
                 :: "v"(loff), "v"(ga) : "memory");
#endif
}

__device__ inline void wait_async_le2() {
#if __has_builtin(__builtin_amdgcn_s_wait_asynccnt)
    __builtin_amdgcn_s_wait_asynccnt(2);
#else
    asm volatile("s_wait_asynccnt 0x2" ::: "memory");
#endif
}
__device__ inline void wait_async_le0() {
#if __has_builtin(__builtin_amdgcn_s_wait_asynccnt)
    __builtin_amdgcn_s_wait_asynccnt(0);
#else
    asm volatile("s_wait_asynccnt 0x0" ::: "memory");
#endif
}

// ---------------- weight convert + transpose: Wt[n*K+k] = bf16(W[k*N+n]) ----------------
__global__ void wt_kernel(const float* __restrict__ W, __bf16* __restrict__ Wt,
                          int K, int N) {
    size_t i = (size_t)blockIdx.x * 256 + threadIdx.x;
    size_t total = (size_t)K * N;
    if (i >= total) return;
    int n = (int)(i / K);
    int k = (int)(i % K);
    Wt[i] = (__bf16)W[(size_t)k * N + n];
}

// ---------------- layernorm -> bf16 ----------------
__global__ void ln_kernel(const float* __restrict__ x, const float* __restrict__ g,
                          const float* __restrict__ be, __bf16* __restrict__ out) {
    int row = blockIdx.x;
    const float* xr = x + (size_t)row * N_EMBD;
    __shared__ float sbuf[256];
    __shared__ float s2buf[256];
    float s = 0.f, ss = 0.f;
    for (int j = threadIdx.x; j < N_EMBD; j += 256) {
        float v = xr[j]; s += v; ss += v * v;
    }
    sbuf[threadIdx.x] = s; s2buf[threadIdx.x] = ss;
    __syncthreads();
    for (int st = 128; st > 0; st >>= 1) {
        if (threadIdx.x < st) {
            sbuf[threadIdx.x]  += sbuf[threadIdx.x + st];
            s2buf[threadIdx.x] += s2buf[threadIdx.x + st];
        }
        __syncthreads();
    }
    float mean = sbuf[0] * (1.0f / N_EMBD);
    float var  = s2buf[0] * (1.0f / N_EMBD) - mean * mean;
    float rs   = rsqrtf(var + 1e-5f);
    for (int j = threadIdx.x; j < N_EMBD; j += 256) {
        out[(size_t)row * N_EMBD + j] = (__bf16)((xr[j] - mean) * rs * g[j] + be[j]);
    }
}

// ---------------- generic bf16 WMMA GEMM with async-LDS B staging ----------------
// C[M,N] = A[M,K](bf16,row) * Bt[N,K](bf16, B transposed) (+bias)(relu?)(+resid)
// block = 128 threads (4 waves); block computes 64x64; B tile (64x32) staged in
// LDS via global_load_async_to_lds_b128, double-buffered across K-steps.
__global__ __launch_bounds__(128) void gemm_kernel(
    const __bf16* __restrict__ A, const __bf16* __restrict__ Bt,
    const float* __restrict__ bias, const float* __restrict__ resid,
    float* __restrict__ outF, __bf16* __restrict__ outH,
    int M, int N, int K, int relu) {
    __shared__ __attribute__((aligned(128))) __bf16 bs[2][64 * 32];

    int tid  = threadIdx.x;
    int lane = tid & 31;
    int wave = tid >> 5;
    int l16 = lane & 15;
    int hi  = lane >> 4;
    int khalf   = hi * 8;
    int khalf16 = hi * 16;

    int arow  = blockIdx.y * 64 + wave * 16 + l16;
    int nbase = blockIdx.x * 64;
    const __bf16* arowp = A + (size_t)arow * K + khalf16;

    // 64x32 bf16 tile = 256 chunks of 16B; 2 chunks per thread.
    int c0 = tid,       col0 = c0 >> 2, part0 = (c0 & 3) * 8;
    int c1 = tid + 128, col1 = c1 >> 2, part1 = (c1 & 3) * 8;
    const __bf16* g0 = Bt + (size_t)(nbase + col0) * K + part0;
    const __bf16* g1 = Bt + (size_t)(nbase + col1) * K + part1;
    int lo0 = col0 * 32 + part0;
    int lo1 = col1 * 32 + part1;

    v8f cacc0 = {}, cacc1 = {}, cacc2 = {}, cacc3 = {};
    int nt = K >> 5;

    async_b128(g0, &bs[0][lo0]);
    async_b128(g1, &bs[0][lo1]);
    for (int kt = 0; kt < nt; ++kt) {
        int k = kt << 5;
        if (kt + 1 < nt) {
            async_b128(g0 + k + 32, &bs[(kt + 1) & 1][lo0]);
            async_b128(g1 + k + 32, &bs[(kt + 1) & 1][lo1]);
            wait_async_le2();   // current tile done (asyncs complete in order)
        } else {
            wait_async_le0();
        }
        __syncthreads();
        v16bf a = ldA(arowp + k);
        const __bf16* bb = &bs[kt & 1][khalf16];
        v16bf b0 = ldB(bb + (0 * 16 + l16) * 32);
        v16bf b1 = ldB(bb + (1 * 16 + l16) * 32);
        v16bf b2 = ldB(bb + (2 * 16 + l16) * 32);
        v16bf b3 = ldB(bb + (3 * 16 + l16) * 32);
        cacc0 = wmma_bf16(a, b0, cacc0);
        cacc1 = wmma_bf16(a, b1, cacc1);
        cacc2 = wmma_bf16(a, b2, cacc2);
        cacc3 = wmma_bf16(a, b3, cacc3);
        __syncthreads();
    }

    int mrow = blockIdx.y * 64 + wave * 16;
#pragma unroll
    for (int r = 0; r < 8; ++r) {
        int row = mrow + r + 8 * hi;
#pragma unroll
        for (int n = 0; n < 4; ++n) {
            int col = nbase + n * 16 + l16;
            float v = (n == 0 ? cacc0[r] : n == 1 ? cacc1[r] : n == 2 ? cacc2[r] : cacc3[r]);
            if (bias)  v += bias[col];
            if (relu)  v = fmaxf(v, 0.f);
            if (resid) v += resid[(size_t)row * N + col];
            size_t idx = (size_t)row * N + col;
            if (outF) outF[idx] = v;
            if (outH) outH[idx] = (__bf16)v;
        }
    }
}

// ---------------- V transpose: [B*T, C] -> [B*H, D, T] ----------------
__global__ void vtr_kernel(const __bf16* __restrict__ V, __bf16* __restrict__ Vt) {
    size_t i = (size_t)blockIdx.x * 256 + threadIdx.x;
    size_t total = (size_t)BT * N_EMBD;
    if (i >= total) return;
    int c  = (int)(i & (N_EMBD - 1));
    size_t bt = i >> 10;
    int b = (int)(bt >> 11);
    int t = (int)(bt & (TT - 1));
    int h = c >> 6, d = c & 63;
    Vt[(((size_t)(b * N_HEAD + h) * HEAD_SZ + d) << 11) + t] = V[i];
}

// ---------------- flash attention (causal, online softmax), 1 wave / 16-query tile ----------------
__global__ __launch_bounds__(32) void attn_kernel(
    const __bf16* __restrict__ Q, const __bf16* __restrict__ Km,
    const __bf16* __restrict__ Vt, __bf16* __restrict__ O) {
    const float scale = 0.03125f;  // C^-0.5 = 1/32 (faithful: uses n_embd)
    int lane = threadIdx.x;
    int l16 = lane & 15;
    int hi  = lane >> 4;
    int khalf   = hi * 8;
    int khalf16 = hi * 16;

    int qt = blockIdx.x;            // query tile
    int bh = blockIdx.y;            // b * H + h
    int b  = bh >> 4;
    int h  = bh & 15;
    int qbase = qt * 16;
    int q = qbase + l16;

    const __bf16* qrow = Q + ((size_t)(b * TT + q)) * N_EMBD + h * HEAD_SZ;
    v16bf aQ0 = ldA(qrow + khalf16);        // d in [0,32)
    v16bf aQ1 = ldA(qrow + 32 + khalf16);   // d in [32,64)

    v8f o0 = {}, o1 = {}, o2 = {}, o3 = {};
    float mrow[8], lrow[8];
#pragma unroll
    for (int r = 0; r < 8; ++r) { mrow[r] = -3.0e38f; lrow[r] = 0.f; }

    __shared__ __bf16 plds[16 * 32];
    const __bf16* vbase = Vt + (size_t)bh * HEAD_SZ * TT;

    for (int kb = 0; kb <= qbase; kb += 32) {
        float s[2][8];
#pragma unroll
        for (int t2 = 0; t2 < 2; ++t2) {
            int kt  = kb + t2 * 16;
            int key = kt + l16;
            const __bf16* krow = Km + ((size_t)(b * TT + key)) * N_EMBD + h * HEAD_SZ;
            v16bf bK0 = ldB(krow + khalf16);
            v16bf bK1 = ldB(krow + 32 + khalf16);
            v8f cs = {};
            cs = wmma_bf16(aQ0, bK0, cs);
            cs = wmma_bf16(aQ1, bK1, cs);
#pragma unroll
            for (int r = 0; r < 8; ++r) {
                float v = cs[r] * scale;
                int qq = qbase + r + 8 * hi;
                if (key > qq) v = -3.0e38f;   // causal mask
                s[t2][r] = v;
            }
        }
        float alpha[8];
#pragma unroll
        for (int r = 0; r < 8; ++r) {
            float mt = fmaxf(s[0][r], s[1][r]);
            mt = fmaxf(mt, __shfl_xor(mt, 1, 16));
            mt = fmaxf(mt, __shfl_xor(mt, 2, 16));
            mt = fmaxf(mt, __shfl_xor(mt, 4, 16));
            mt = fmaxf(mt, __shfl_xor(mt, 8, 16));
            float mn = fmaxf(mrow[r], mt);
            alpha[r] = __expf(mrow[r] - mn);
            mrow[r] = mn;
            float p0 = __expf(s[0][r] - mn);
            float p1 = __expf(s[1][r] - mn);
            s[0][r] = p0; s[1][r] = p1;
            float rs = p0 + p1;
            rs += __shfl_xor(rs, 1, 16);
            rs += __shfl_xor(rs, 2, 16);
            rs += __shfl_xor(rs, 4, 16);
            rs += __shfl_xor(rs, 8, 16);
            lrow[r] = lrow[r] * alpha[r] + rs;
        }
#pragma unroll
        for (int r = 0; r < 8; ++r) {
            o0[r] *= alpha[r]; o1[r] *= alpha[r];
            o2[r] *= alpha[r]; o3[r] *= alpha[r];
        }
        // C-layout -> A-layout bounce through LDS
#pragma unroll
        for (int r = 0; r < 8; ++r) {
            int M = r + 8 * hi;
            plds[M * 32 +      l16] = (__bf16)s[0][r];
            plds[M * 32 + 16 + l16] = (__bf16)s[1][r];
        }
        __syncthreads();
        v16bf aP = ldA(&plds[l16 * 32 + khalf16]);
        __syncthreads();
        // O += P(16x32) @ V(32x64), 4 n-tiles; Vt rows contiguous in key
#pragma unroll
        for (int n = 0; n < 4; ++n) {
            int d = n * 16 + l16;
            v16bf bV = ldB(vbase + (size_t)d * TT + kb + khalf16);
            if (n == 0) o0 = wmma_bf16(aP, bV, o0);
            else if (n == 1) o1 = wmma_bf16(aP, bV, o1);
            else if (n == 2) o2 = wmma_bf16(aP, bV, o2);
            else o3 = wmma_bf16(aP, bV, o3);
        }
    }
#pragma unroll
    for (int r = 0; r < 8; ++r) {
        float inv = 1.0f / lrow[r];
        size_t row = (size_t)(b * TT + qbase + r + 8 * hi);
        __bf16* orow = O + row * N_EMBD + h * HEAD_SZ + l16;
        orow[0]  = (__bf16)(o0[r] * inv);
        orow[16] = (__bf16)(o1[r] * inv);
        orow[32] = (__bf16)(o2[r] * inv);
        orow[48] = (__bf16)(o3[r] * inv);
    }
}

// ---------------- launcher ----------------
extern "C" void kernel_launch(void* const* d_in, const int* in_sizes, int n_in,
                              void* d_out, int out_size, void* d_ws, size_t ws_size,
                              hipStream_t stream) {
    (void)in_sizes; (void)n_in; (void)out_size; (void)ws_size;
    const float* x   = (const float*)d_in[0];
    const float* Wq  = (const float*)d_in[1];
    const float* Wk  = (const float*)d_in[2];
    const float* Wv  = (const float*)d_in[3];
    const float* Wo  = (const float*)d_in[4];
    const float* bo  = (const float*)d_in[5];
    const float* W1  = (const float*)d_in[6];
    const float* b1  = (const float*)d_in[7];
    const float* W2  = (const float*)d_in[8];
    const float* b2  = (const float*)d_in[9];
    const float* g1  = (const float*)d_in[10];
    const float* be1 = (const float*)d_in[11];
    const float* g2  = (const float*)d_in[12];
    const float* be2 = (const float*)d_in[13];
    float* out = (float*)d_out;

    // workspace carve-up (256B aligned)
    char* ws = (char*)d_ws;
    size_t off = 0;
    auto alloc = [&](size_t bytes) -> char* {
        char* p = ws + off;
        off = (off + bytes + 255) & ~(size_t)255;
        return p;
    };
    __bf16* WqT  = (__bf16*)alloc((size_t)N_EMBD * N_EMBD * 2);
    __bf16* WkT  = (__bf16*)alloc((size_t)N_EMBD * N_EMBD * 2);
    __bf16* WvT  = (__bf16*)alloc((size_t)N_EMBD * N_EMBD * 2);
    __bf16* WoT  = (__bf16*)alloc((size_t)N_EMBD * N_EMBD * 2);
    __bf16* W1T  = (__bf16*)alloc((size_t)N_EMBD * FF * 2);
    __bf16* W2T  = (__bf16*)alloc((size_t)FF * N_EMBD * 2);
    __bf16* hln  = (__bf16*)alloc((size_t)BT * N_EMBD * 2);   // reused for ln1 and ln2
    __bf16* Qb   = (__bf16*)alloc((size_t)BT * N_EMBD * 2);
    __bf16* Kb   = (__bf16*)alloc((size_t)BT * N_EMBD * 2);
    __bf16* Vb   = (__bf16*)alloc((size_t)BT * N_EMBD * 2);
    __bf16* Vtb  = (__bf16*)alloc((size_t)BT * N_EMBD * 2);
    __bf16* attb = (__bf16*)alloc((size_t)BT * N_EMBD * 2);
    float*  x1   = (float*)alloc((size_t)BT * N_EMBD * 4);
    __bf16* ffb  = (__bf16*)alloc((size_t)BT * FF * 2);

    const int TPB = 256;
    size_t e1 = (size_t)N_EMBD * N_EMBD;
    size_t e4 = (size_t)N_EMBD * FF;
    wt_kernel<<<(unsigned)((e1 + TPB - 1) / TPB), TPB, 0, stream>>>(Wq, WqT, N_EMBD, N_EMBD);
    wt_kernel<<<(unsigned)((e1 + TPB - 1) / TPB), TPB, 0, stream>>>(Wk, WkT, N_EMBD, N_EMBD);
    wt_kernel<<<(unsigned)((e1 + TPB - 1) / TPB), TPB, 0, stream>>>(Wv, WvT, N_EMBD, N_EMBD);
    wt_kernel<<<(unsigned)((e1 + TPB - 1) / TPB), TPB, 0, stream>>>(Wo, WoT, N_EMBD, N_EMBD);
    wt_kernel<<<(unsigned)((e4 + TPB - 1) / TPB), TPB, 0, stream>>>(W1, W1T, N_EMBD, FF);
    wt_kernel<<<(unsigned)((e4 + TPB - 1) / TPB), TPB, 0, stream>>>(W2, W2T, FF, N_EMBD);

    ln_kernel<<<BT, 256, 0, stream>>>(x, g1, be1, hln);

    dim3 gqkv(N_EMBD / 64, BT / 64);
    gemm_kernel<<<gqkv, 128, 0, stream>>>(hln, WqT, nullptr, nullptr, nullptr, Qb,
                                          BT, N_EMBD, N_EMBD, 0);
    gemm_kernel<<<gqkv, 128, 0, stream>>>(hln, WkT, nullptr, nullptr, nullptr, Kb,
                                          BT, N_EMBD, N_EMBD, 0);
    gemm_kernel<<<gqkv, 128, 0, stream>>>(hln, WvT, nullptr, nullptr, nullptr, Vb,
                                          BT, N_EMBD, N_EMBD, 0);

    size_t ev = (size_t)BT * N_EMBD;
    vtr_kernel<<<(unsigned)((ev + TPB - 1) / TPB), TPB, 0, stream>>>(Vb, Vtb);

    attn_kernel<<<dim3(TT / 16, BB * N_HEAD), 32, 0, stream>>>(Qb, Kb, Vtb, attb);

    gemm_kernel<<<gqkv, 128, 0, stream>>>(attb, WoT, bo, x, x1, nullptr,
                                          BT, N_EMBD, N_EMBD, 0);

    ln_kernel<<<BT, 256, 0, stream>>>(x1, g2, be2, hln);

    gemm_kernel<<<dim3(FF / 64, BT / 64), 128, 0, stream>>>(hln, W1T, b1, nullptr,
                                          nullptr, ffb, BT, FF, N_EMBD, 1);

    gemm_kernel<<<gqkv, 128, 0, stream>>>(ffb, W2T, b2, x1, out, nullptr,
                                          BT, N_EMBD, FF, 0);
}
